// MultiHeadAttention_45956150067200
// MI455X (gfx1250) — matrix operand order repaired and checked by reading, not compiled, not run on hardware
//
#include <hip/hip_runtime.h>
#include <hip/hip_bf16.h>
#include <math.h>

// MHA: b=2, seq(m)=seq(n)=2048, e=128, heads k=8, head_dim d=16.
// Fused flash-attention pipeline, all GEMMs via v_wmma_f32_16x16x32_f16.

typedef _Float16 v16h __attribute__((ext_vector_type(16)));
typedef _Float16 v8h  __attribute__((ext_vector_type(8)));
typedef float    v8f  __attribute__((ext_vector_type(8)));

constexpr int BATCH = 2;
constexpr int SEQ   = 2048;   // both key (m) and query (n) length
constexpr int EMB   = 128;
constexpr int NHEAD = 8;
constexpr int HDIM  = 16;
// fold head_dim^-0.5 * log2(e) into Q so softmax uses exp2 (v_exp_f32)
#define QSCALE 0.3606737602222409f

__device__ __forceinline__ v8f wmma16(v16h a, v16h b, v8f c) {
  return __builtin_amdgcn_wmma_f32_16x16x32_f16(false, a, false, b, (short)0, c,
                                                false, false);
}

// A-operand 16x32 f16 tile from a row-major matrix (row stride ld halves).
// Layout used here (contiguous k): lanes 0-15 hold M=lane, K=0..15; lanes 16-31 hold
// M=lane-16, K=16..31  -> two contiguous 16B loads per lane.
__device__ __forceinline__ v16h load_a(const _Float16* tile, int ld, int lane) {
  int r = lane & 15, g = lane >> 4;
  const _Float16* p = tile + r * ld + g * 16;
  v8h lo = *(const v8h*)p;
  v8h hi = *(const v8h*)(p + 8);
  v16h a;
#pragma unroll
  for (int i = 0; i < 8; ++i) { a[i] = lo[i]; a[8 + i] = hi[i]; }
  return a;
}

// B-operand 32x16 f16 tile; source stored TRANSPOSED [N][K] row-major
// (stride ld). Layout used here (contiguous k): lane -> N = lane%16, K = (lane/16)*16 .. +15,
// i.e. one contiguous 32B load per lane.
__device__ __forceinline__ v16h load_bt(const _Float16* tileT, int ld, int lane) {
  int n = lane & 15, g = lane >> 4;
  return *(const v16h*)(tileT + n * ld + g * 16);
}

// ---------------------------------------------------------------- pack ----
// Convert x,y to f16; pack weights transposed: Wt[k][d][e] so B-operand
// loads are contiguous; W2t[q][k*16+d] for the output projection.
__global__ void mha_pack(const float* __restrict__ x, const float* __restrict__ y,
                         const float* __restrict__ l1, const float* __restrict__ l2,
                         const float* __restrict__ t1, const float* __restrict__ t2,
                         _Float16* __restrict__ xh, _Float16* __restrict__ yh,
                         _Float16* __restrict__ L1t, _Float16* __restrict__ L2t,
                         _Float16* __restrict__ T1t, _Float16* __restrict__ W2t) {
  int tid = blockIdx.x * blockDim.x + threadIdx.x;
  int nth = gridDim.x * blockDim.x;
  const int NX = BATCH * SEQ * EMB;
  for (int i = tid; i < NX; i += nth) {
    xh[i] = (_Float16)x[i];
    yh[i] = (_Float16)y[i];
  }
  const int NW = NHEAD * HDIM * EMB;  // [k][d][e] <- [k][e][d]
  for (int i = tid; i < NW; i += nth) {
    int k = i / (HDIM * EMB);
    int rem = i - k * HDIM * EMB;
    int d = rem / EMB;
    int e = rem - d * EMB;
    int src = (k * EMB + e) * HDIM + d;
    L1t[i] = (_Float16)l1[src];
    L2t[i] = (_Float16)l2[src];
    T1t[i] = (_Float16)t1[src];
  }
  const int NW2 = EMB * NHEAD * HDIM;  // W2t[q][k*16+d] <- theta2[k][q][d]
  for (int i = tid; i < NW2; i += nth) {
    int q = i / (NHEAD * HDIM);
    int c = i - q * NHEAD * HDIM;
    int k = c / HDIM, d = c - k * HDIM;
    W2t[i] = (_Float16)t2[(k * EMB + q) * HDIM + d];
  }
}

// ------------------------------------------------------------ QKV proj ----
// One wave per (b, head, 16-row tile): three K=128 GEMMs (4 WMMAs each).
// Q <- y*lambda2 (pre-scaled), K <- x*lambda1 + bias_lambda, V <- x*theta1
// (stored transposed [d][m] for the PV B-operand).
__global__ void mha_qkv(const _Float16* __restrict__ xh, const _Float16* __restrict__ yh,
                        const _Float16* __restrict__ L1t, const _Float16* __restrict__ L2t,
                        const _Float16* __restrict__ T1t,
                        const float* __restrict__ bias_lambda,
                        _Float16* __restrict__ Qh, _Float16* __restrict__ Kh,
                        _Float16* __restrict__ Vt) {
  int lane = threadIdx.x & 31;
  int w = blockIdx.x * (blockDim.x >> 5) + (threadIdx.x >> 5);  // 0..2047
  int t = w & 127;
  int k = (w >> 7) & 7;
  int b = w >> 10;

  v8f accQ, accK, accV;
#pragma unroll
  for (int i = 0; i < 8; ++i) { accQ[i] = 0.f; accK[i] = 0.f; accV[i] = 0.f; }

  const _Float16* xrow = xh + (b * SEQ + t * 16) * EMB;
  const _Float16* yrow = yh + (b * SEQ + t * 16) * EMB;
  const _Float16* l1k = L1t + k * HDIM * EMB;
  const _Float16* l2k = L2t + k * HDIM * EMB;
  const _Float16* t1k = T1t + k * HDIM * EMB;
#pragma unroll
  for (int c = 0; c < 4; ++c) {
    v16h ax = load_a(xrow + c * 32, EMB, lane);
    v16h ay = load_a(yrow + c * 32, EMB, lane);
    v16h b1 = load_bt(l1k + c * 32, EMB, lane);
    v16h b2 = load_bt(l2k + c * 32, EMB, lane);
    v16h b3 = load_bt(t1k + c * 32, EMB, lane);
    accK = wmma16(ax, b1, accK);
    accQ = wmma16(ay, b2, accQ);
    accV = wmma16(ax, b3, accV);
  }
  int n = lane & 15, g = lane >> 4;
  int head = b * NHEAD + k;
  float bl = bias_lambda[k * HDIM + n];
#pragma unroll
  for (int r = 0; r < 8; ++r) {
    int row = t * 16 + r + 8 * g;
    Qh[(head * SEQ + row) * HDIM + n] = (_Float16)(accQ[r] * QSCALE);
    Kh[(head * SEQ + row) * HDIM + n] = (_Float16)(accK[r] + bl);
    Vt[(head * HDIM + n) * SEQ + row] = (_Float16)accV[r];
  }
}

// ------------------------------------------------------- flash attention --
// One wave per (b, head, 16-query tile); online softmax over all 2048 keys
// in 64-key rounds: 4 WMMAs for S (d padded 16->32), ONE max-reduce and ONE
// sum-reduce per row per round (amortized), LDS transpose of P (C-layout ->
// A-layout), 2 WMMAs for O += P*V.
__global__ void mha_flash(const _Float16* __restrict__ Qh, const _Float16* __restrict__ Kh,
                          const _Float16* __restrict__ Vt,
                          const float* __restrict__ bias_theta,
                          _Float16* __restrict__ Oh) {
  __shared__ __attribute__((aligned(32))) _Float16 lds[4][16 * 64];
  int lane = threadIdx.x & 31;
  int wave = threadIdx.x >> 5;
  int w = blockIdx.x * (blockDim.x >> 5) + wave;  // 0..2047
  int nt = w & 127;
  int k = (w >> 7) & 7;
  int b = w >> 10;
  int head = b * NHEAD + k;

  const _Float16* Qbase = Qh + (head * SEQ + nt * 16) * HDIM;
  const _Float16* Kbase = Kh + head * SEQ * HDIM;
  const _Float16* Vbase = Vt + head * HDIM * SEQ;
  _Float16* P = lds[wave];  // 16 x 64 f16, row stride 64

  int n = lane & 15, g = lane >> 4;

  // Q tile as A-operand, K-dim zero-padded 16 -> 32
  v16h aq;
  {
    v8h lo = *(const v8h*)(Qbase + n * HDIM);
    v8h hi = *(const v8h*)(Qbase + n * HDIM + 8);
#pragma unroll
    for (int i = 0; i < 8; ++i) { aq[i] = g ? (_Float16)0.f : lo[i]; aq[8 + i] = g ? (_Float16)0.f : hi[i]; }
  }

  v8f zero;
#pragma unroll
  for (int i = 0; i < 8; ++i) zero[i] = 0.f;

  v8f acc;
  float mrow[8], lrow[8];
#pragma unroll
  for (int r = 0; r < 8; ++r) {
    acc[r] = 0.f;
    mrow[r] = -__builtin_inff();
    lrow[r] = 0.f;
  }

  for (int m0 = 0; m0 < SEQ; m0 += 64) {
    // 4 key tiles as B-operand: lanes 0-15 carry K=d (real), 16-31 pad 0
    v16h bk[4];
#pragma unroll
    for (int c = 0; c < 4; ++c) {
#pragma unroll
      for (int i = 0; i < 16; ++i) bk[c][i] = (_Float16)0.f;
    }
    if (lane < 16) {
#pragma unroll
      for (int c = 0; c < 4; ++c)
        bk[c] = *(const v16h*)(Kbase + (m0 + c * 16 + lane) * HDIM);
    }
    v8f s0 = wmma16(aq, bk[0], zero);  // queries x keys[m0 +  0..15]
    v8f s1 = wmma16(aq, bk[1], zero);  // queries x keys[m0 + 16..31]
    v8f s2 = wmma16(aq, bk[2], zero);  // queries x keys[m0 + 32..47]
    v8f s3 = wmma16(aq, bk[3], zero);  // queries x keys[m0 + 48..63]

    // Online softmax in the C/D layout: lane owns 8 rows (M = r + 8*g),
    // the 16 columns of a row live in lanes sharing the same half-wave.
    // One cross-lane max + one cross-lane add per row per 64 keys.
#pragma unroll
    for (int r = 0; r < 8; ++r) {
      float mx = fmaxf(fmaxf(s0[r], s1[r]), fmaxf(s2[r], s3[r]));
#pragma unroll
      for (int s = 1; s < 16; s <<= 1) mx = fmaxf(mx, __shfl_xor(mx, s, 32));
      float mnew = fmaxf(mrow[r], mx);
      float p0 = exp2f(s0[r] - mnew);
      float p1 = exp2f(s1[r] - mnew);
      float p2 = exp2f(s2[r] - mnew);
      float p3 = exp2f(s3[r] - mnew);
      float corr = exp2f(mrow[r] - mnew);
      float rs = (p0 + p1) + (p2 + p3);
#pragma unroll
      for (int s = 1; s < 16; s <<= 1) rs += __shfl_xor(rs, s, 32);
      lrow[r] = lrow[r] * corr + rs;
      mrow[r] = mnew;
      acc[r] *= corr;
      int row = r + 8 * g;
      P[row * 64 + n]      = (_Float16)p0;  // key column m0 + n
      P[row * 64 + 16 + n] = (_Float16)p1;  // key column m0 + 16 + n
      P[row * 64 + 32 + n] = (_Float16)p2;  // key column m0 + 32 + n
      P[row * 64 + 48 + n] = (_Float16)p3;  // key column m0 + 48 + n
    }
    // wave-private LDS region; DS ops are in-order per wave
    asm volatile("s_wait_dscnt 0" ::: "memory");
    v16h ap0 = load_a(P, 64, lane);
    v16h ap1 = load_a(P + 32, 64, lane);
    v16h bv0 = load_bt(Vbase + m0, SEQ, lane);       // Vt[d][m0 + 0..31]
    v16h bv1 = load_bt(Vbase + m0 + 32, SEQ, lane);  // Vt[d][m0 + 32..63]
    acc = wmma16(ap0, bv0, acc);
    acc = wmma16(ap1, bv1, acc);
  }

  float bt = bias_theta[k * HDIM + n];
#pragma unroll
  for (int r = 0; r < 8; ++r) {
    float o = acc[r] / lrow[r] + bt;
    int row = nt * 16 + r + 8 * g;
    Oh[(b * SEQ + row) * EMB + k * HDIM + n] = (_Float16)o;
  }
}

// ------------------------------------------------------ output project ----
// out[bn][q] = Oh[bn][kd] * W2t[q][kd]^T : [4096x128] x [128x128], f32 out.
__global__ void mha_outproj(const _Float16* __restrict__ Oh,
                            const _Float16* __restrict__ W2t,
                            float* __restrict__ out) {
  int lane = threadIdx.x & 31;
  int w = blockIdx.x * (blockDim.x >> 5) + (threadIdx.x >> 5);  // 0..2047
  int qt = w & 7;    // 8 column tiles of 16
  int rt = w >> 3;   // 256 row tiles of 16
  const _Float16* Arow = Oh + rt * 16 * EMB;
  const _Float16* Bcol = W2t + qt * 16 * EMB;
  v8f acc;
#pragma unroll
  for (int i = 0; i < 8; ++i) acc[i] = 0.f;
#pragma unroll
  for (int c = 0; c < 4; ++c) {
    v16h a = load_a(Arow + c * 32, EMB, lane);
    v16h bt = load_bt(Bcol + c * 32, EMB, lane);
    acc = wmma16(a, bt, acc);
  }
  int n = lane & 15, g = lane >> 4;
#pragma unroll
  for (int r = 0; r < 8; ++r) {
    int row = rt * 16 + r + 8 * g;
    out[row * EMB + qt * 16 + n] = acc[r];
  }
}

extern "C" void kernel_launch(void* const* d_in, const int* in_sizes, int n_in,
                              void* d_out, int out_size, void* d_ws, size_t ws_size,
                              hipStream_t stream) {
  (void)in_sizes; (void)n_in; (void)out_size; (void)ws_size;
  const float* x  = (const float*)d_in[0];
  const float* y  = (const float*)d_in[1];
  const float* l1 = (const float*)d_in[2];
  const float* l2 = (const float*)d_in[3];
  const float* t1 = (const float*)d_in[4];
  const float* t2 = (const float*)d_in[5];
  const float* bl = (const float*)d_in[6];
  const float* bt = (const float*)d_in[7];

  char* ws = (char*)d_ws;
  size_t off = 0;
  auto grab = [&](size_t bytes) -> char* {
    char* p = ws + off;
    off += (bytes + 255) & ~(size_t)255;
    return p;
  };
  _Float16* xh  = (_Float16*)grab((size_t)BATCH * SEQ * EMB * 2);
  _Float16* yh  = (_Float16*)grab((size_t)BATCH * SEQ * EMB * 2);
  _Float16* L1t = (_Float16*)grab((size_t)NHEAD * HDIM * EMB * 2);
  _Float16* L2t = (_Float16*)grab((size_t)NHEAD * HDIM * EMB * 2);
  _Float16* T1t = (_Float16*)grab((size_t)NHEAD * HDIM * EMB * 2);
  _Float16* W2t = (_Float16*)grab((size_t)EMB * NHEAD * HDIM * 2);
  _Float16* Qh  = (_Float16*)grab((size_t)BATCH * NHEAD * SEQ * HDIM * 2);
  _Float16* Kh  = (_Float16*)grab((size_t)BATCH * NHEAD * SEQ * HDIM * 2);
  _Float16* Vt  = (_Float16*)grab((size_t)BATCH * NHEAD * HDIM * SEQ * 2);
  _Float16* Oh  = (_Float16*)grab((size_t)BATCH * SEQ * EMB * 2);

  mha_pack<<<512, 256, 0, stream>>>(x, y, l1, l2, t1, t2, xh, yh, L1t, L2t, T1t, W2t);
  mha_qkv<<<256, 256, 0, stream>>>(xh, yh, L1t, L2t, T1t, bl, Qh, Kh, Vt);
  mha_flash<<<512, 128, 0, stream>>>(Qh, Kh, Vt, bt, Oh);
  mha_outproj<<<256, 256, 0, stream>>>(Oh, W2t, (float*)d_out);
}
